// SSMBlock_1958505087212
// MI455X (gfx1250) — matrix operand order repaired and checked by reading, not compiled, not run on hardware
//
#include <hip/hip_runtime.h>
#include <math.h>

// ---------------------------------------------------------------------------
// Problem constants
// ---------------------------------------------------------------------------
#define BATCH   4
#define SEQLEN  2048
#define DMODEL  1024
#define NSTATE  16
#define DFF     4096
#define TOKENS  (BATCH * SEQLEN)      // 8192
#define NCHUNK  32
#define CHUNKL  (SEQLEN / NCHUNK)     // 64

typedef __attribute__((ext_vector_type(16))) __bf16 v16bf;
typedef __attribute__((ext_vector_type(8)))  float  v8f;

// float -> bf16 (round to nearest even)
__device__ __forceinline__ unsigned short f2bf(float f) {
    unsigned int u = __float_as_uint(f);
    unsigned int r = u + 0x7fffu + ((u >> 16) & 1u);
    return (unsigned short)(r >> 16);
}

// ---------------------------------------------------------------------------
// K0: convert fp32 weight [K][N] -> bf16 transposed [N][K]
// ---------------------------------------------------------------------------
__global__ void conv_transpose_kernel(const float* __restrict__ w,
                                      unsigned short* __restrict__ wt,
                                      int K, int N) {
    size_t idx = (size_t)blockIdx.x * 256 + threadIdx.x;
    int n = (int)(idx % N);
    int k = (int)(idx / N);
    wt[(size_t)n * K + k] = f2bf(w[idx]);
}

// ---------------------------------------------------------------------------
// RMSNorm: one block per token, 256 threads x 4 elements (D=1024)
// OUT_BF16=0 -> fp32 out; OUT_BF16=1 -> packed bf16 out
// ---------------------------------------------------------------------------
template <int OUT_BF16>
__global__ void rmsnorm_kernel(const float* __restrict__ x,
                               const float* __restrict__ scale,
                               float* __restrict__ outf,
                               unsigned short* __restrict__ outb) {
    int token = blockIdx.x;
    int tid   = threadIdx.x;
    const float4 v = *reinterpret_cast<const float4*>(x + (size_t)token * DMODEL + tid * 4);
    float ss = v.x * v.x + v.y * v.y + v.z * v.z + v.w * v.w;
#pragma unroll
    for (int off = 16; off > 0; off >>= 1) ss += __shfl_xor(ss, off, 32);
    __shared__ float sred[8];
    int wid = tid >> 5, lane = tid & 31;
    if (lane == 0) sred[wid] = ss;
    __syncthreads();
    if (tid == 0) {
        float t = 0.f;
#pragma unroll
        for (int i = 0; i < 8; i++) t += sred[i];
        sred[0] = t;
    }
    __syncthreads();
    float r = rsqrtf(sred[0] * (1.0f / (float)DMODEL) + 1e-6f);
    const float4 sc = *reinterpret_cast<const float4*>(scale + tid * 4);
    float o0 = v.x * r * sc.x, o1 = v.y * r * sc.y, o2 = v.z * r * sc.z, o3 = v.w * r * sc.w;
    if (OUT_BF16) {
        uint2 pk;
        pk.x = (unsigned)f2bf(o0) | ((unsigned)f2bf(o1) << 16);
        pk.y = (unsigned)f2bf(o2) | ((unsigned)f2bf(o3) << 16);
        *reinterpret_cast<uint2*>(outb + (size_t)token * DMODEL + tid * 4) = pk;
    } else {
        float4 o = make_float4(o0, o1, o2, o3);
        *reinterpret_cast<float4*>(outf + (size_t)token * DMODEL + tid * 4) = o;
    }
}

// ---------------------------------------------------------------------------
// SSM scan, chunked parallel form.  dA is time-invariant, so:
//   s_t = dA^(t-t0+1) * S_in(chunk) + s_local_t
// Phase 1: per-(b,d,chunk) local scan with s=0; emit y_local and chunk-end state.
// Phase 2: per-(b,d,n) prefix over chunks with dA^64.
// Phase 3: correction y += C . (dA^(t-t0+1) S_in), fused with residual add.
// ---------------------------------------------------------------------------
#define LN_DT0  (-6.907755279f)   // ln(0.001)
#define LN_STEP (0.3070113458f)   // ln(100)/15

__device__ __forceinline__ float decay_dA(float a_raw, int n, float* ea_out) {
    float ea = expf(a_raw);
    float dt = expf(LN_DT0 + LN_STEP * (float)n);
    *ea_out  = ea;
    return expf(-ea * dt);
}

__global__ void scan_phase1_kernel(const float* __restrict__ U,
                                   const float* __restrict__ A_ssm,
                                   const float* __restrict__ B_ssm,
                                   const float* __restrict__ C_ssm,
                                   float* __restrict__ Y,
                                   float* __restrict__ Send) {
    int tid   = threadIdx.x;
    int dblk  = blockIdx.x & 3;
    int chunk = (blockIdx.x >> 2) & (NCHUNK - 1);
    int b     = blockIdx.x >> 7;
    int d     = dblk * 256 + tid;

    float dA[NSTATE], dB[NSTATE], Cc[NSTATE], s[NSTATE];
#pragma unroll
    for (int n = 0; n < NSTATE; n++) {
        float ea;
        dA[n] = decay_dA(A_ssm[d * NSTATE + n], n, &ea);
        dB[n] = B_ssm[d * NSTATE + n] * (1.0f - dA[n]) / ea;
        Cc[n] = C_ssm[d * NSTATE + n];
        s[n]  = 0.f;
    }
    size_t base = ((size_t)b * SEQLEN + (size_t)chunk * CHUNKL) * DMODEL + d;
    for (int t = 0; t < CHUNKL; t++) {
        float u = U[base + (size_t)t * DMODEL];
        float y = 0.f;
#pragma unroll
        for (int n = 0; n < NSTATE; n++) {
            s[n] = fmaf(s[n], dA[n], u * dB[n]);
            y    = fmaf(Cc[n], s[n], y);
        }
        Y[base + (size_t)t * DMODEL] = y;
    }
    size_t sb = (((size_t)b * NCHUNK + chunk) * NSTATE) * DMODEL + d;
#pragma unroll
    for (int n = 0; n < NSTATE; n++) Send[sb + (size_t)n * DMODEL] = s[n];
}

__global__ void scan_phase2_kernel(const float* __restrict__ A_ssm,
                                   const float* __restrict__ Send,
                                   float* __restrict__ Sin) {
    int idx = blockIdx.x * 256 + threadIdx.x;   // B*N*D = 65536
    int d = idx & (DMODEL - 1);
    int n = (idx >> 10) & (NSTATE - 1);
    int b = idx >> 14;
    float ea;
    float dA = decay_dA(A_ssm[d * NSTATE + n], n, &ea);
    float p = dA;                               // dA^64 via 6 squarings
#pragma unroll
    for (int i = 0; i < 6; i++) p = p * p;
    float S = 0.f;
    for (int c = 0; c < NCHUNK; c++) {
        size_t o = (((size_t)b * NCHUNK + c) * NSTATE + n) * DMODEL + d;
        Sin[o] = S;
        S = fmaf(p, S, Send[o]);
    }
}

__global__ void scan_phase3_kernel(const float* __restrict__ X,
                                   const float* __restrict__ U,
                                   const float* __restrict__ Y,
                                   const float* __restrict__ Sin,
                                   const float* __restrict__ A_ssm,
                                   const float* __restrict__ C_ssm,
                                   const float* __restrict__ D_ssm,
                                   float* __restrict__ X2) {
    int tid   = threadIdx.x;
    int dblk  = blockIdx.x & 3;
    int chunk = (blockIdx.x >> 2) & (NCHUNK - 1);
    int b     = blockIdx.x >> 7;
    int d     = dblk * 256 + tid;

    float dA[NSTATE], Cc[NSTATE], q[NSTATE];
    size_t sb = (((size_t)b * NCHUNK + chunk) * NSTATE) * DMODEL + d;
#pragma unroll
    for (int n = 0; n < NSTATE; n++) {
        float ea;
        dA[n] = decay_dA(A_ssm[d * NSTATE + n], n, &ea);
        Cc[n] = C_ssm[d * NSTATE + n];
        q[n]  = Sin[sb + (size_t)n * DMODEL];
    }
    float Dd = D_ssm[d];
    size_t base = ((size_t)b * SEQLEN + (size_t)chunk * CHUNKL) * DMODEL + d;
    for (int t = 0; t < CHUNKL; t++) {
        float ycorr = 0.f;
#pragma unroll
        for (int n = 0; n < NSTATE; n++) {
            q[n] *= dA[n];
            ycorr = fmaf(Cc[n], q[n], ycorr);
        }
        size_t o = base + (size_t)t * DMODEL;
        X2[o] = X[o] + Y[o] + ycorr + U[o] * Dd;
    }
}

// ---------------------------------------------------------------------------
// bf16 WMMA GEMM:  C[M,N] = A[M,K] * W[K,N], W supplied pre-transposed [N][K].
// 256 threads = 8 waves (2x4); BM=BN=128, BK=32; each wave 64x32 output
// (4x2 tiles of 16x16 via v_wmma_f32_16x16x32_bf16).
// MODE 0: +bias, exact GELU, store bf16.   MODE 1: +bias +residual, store f32.
// ---------------------------------------------------------------------------
template <int MODE>
__global__ void __launch_bounds__(256)
gemm_bf16_kernel(const unsigned short* __restrict__ A,   // [M][K] bf16
                 const unsigned short* __restrict__ Bt,  // [N][K] bf16 (W^T)
                 const float* __restrict__ bias,         // [N]
                 const float* __restrict__ res,          // [M][N] (MODE 1)
                 void* __restrict__ outp,
                 int M, int N, int K) {
    constexpr int BM = 128, BN = 128, BK = 32, LDT = 40;  // LDT: padded halfs
    __shared__ __align__(16) unsigned short lds_a[BM * LDT];
    __shared__ __align__(16) unsigned short lds_b[BN * LDT];

    int tid  = threadIdx.x;
    int lane = tid & 31, wid = tid >> 5;
    int m0 = blockIdx.x * BM, n0 = blockIdx.y * BN;
    int wm = (wid >> 2) * 64, wn = (wid & 3) * 32;
    int lrow = lane & 15, hi = lane >> 4;

    v8f acc[4][2];
    v8f zero = {0.f, 0.f, 0.f, 0.f, 0.f, 0.f, 0.f, 0.f};
#pragma unroll
    for (int mt = 0; mt < 4; mt++)
#pragma unroll
        for (int nt = 0; nt < 2; nt++) acc[mt][nt] = zero;

    int arow = tid >> 1;            // 0..127
    int acol = (tid & 1) * 16;      // 0 or 16 halfs
    const int kiters = K / BK;

    for (int kt = 0; kt < kiters; ++kt) {
        int kbase = kt * BK;
        {   // global -> LDS (16 halfs = 32B per thread per tile)
            const uint4* ga = reinterpret_cast<const uint4*>(
                A + (size_t)(m0 + arow) * K + kbase + acol);
            uint4 a0 = ga[0], a1 = ga[1];
            const uint4* gb = reinterpret_cast<const uint4*>(
                Bt + (size_t)(n0 + arow) * K + kbase + acol);
            uint4 b0 = gb[0], b1 = gb[1];
            uint4* da = reinterpret_cast<uint4*>(&lds_a[arow * LDT + acol]);
            da[0] = a0; da[1] = a1;
            uint4* db = reinterpret_cast<uint4*>(&lds_b[arow * LDT + acol]);
            db[0] = b0; db[1] = b1;
        }
        __syncthreads();

        union Frag { uint4 u[2]; v16bf v; };
        Frag fa[4], fb[2];
        // A frag (16x32): lane<16 -> row lrow, K {0..7,16..23}; lane>=16 -> K {8..15,24..31}
#pragma unroll
        for (int mt = 0; mt < 4; mt++) {
            const unsigned short* p = &lds_a[(wm + mt * 16 + lrow) * LDT + hi * 8];
            fa[mt].u[0] = *reinterpret_cast<const uint4*>(p);
            fa[mt].u[1] = *reinterpret_cast<const uint4*>(p + 16);
        }
        // B frag (32x16) from W^T rows: lane<16 -> col lrow, K {0..7,16..23}; lane>=16 -> K {8..15,24..31}
#pragma unroll
        for (int nt = 0; nt < 2; nt++) {
            const unsigned short* p = &lds_b[(wn + nt * 16 + lrow) * LDT + hi * 8];
            fb[nt].u[0] = *reinterpret_cast<const uint4*>(p);
            fb[nt].u[1] = *reinterpret_cast<const uint4*>(p + 16);
        }
#pragma unroll
        for (int mt = 0; mt < 4; mt++)
#pragma unroll
            for (int nt = 0; nt < 2; nt++)
                acc[mt][nt] = __builtin_amdgcn_wmma_f32_16x16x32_bf16(
                    false, fa[mt].v, false, fb[nt].v, (short)0, acc[mt][nt],
                    false, false);
        __syncthreads();
    }

    // Epilogue. C layout: lanes 0-15: N=lane, M=r; lanes 16-31: N=lane-16, M=r+8.
#pragma unroll
    for (int mt = 0; mt < 4; mt++) {
#pragma unroll
        for (int nt = 0; nt < 2; nt++) {
            int n = n0 + wn + nt * 16 + lrow;
            float bval = bias[n];
#pragma unroll
            for (int r = 0; r < 8; r++) {
                int m = m0 + wm + mt * 16 + hi * 8 + r;
                float v = acc[mt][nt][r] + bval;
                if (MODE == 0) {
                    float g = 0.5f * v * (1.0f + erff(v * 0.70710678118f));
                    reinterpret_cast<unsigned short*>(outp)[(size_t)m * N + n] = f2bf(g);
                } else {
                    reinterpret_cast<float*>(outp)[(size_t)m * N + n] =
                        v + res[(size_t)m * N + n];
                }
            }
        }
    }
}

// ---------------------------------------------------------------------------
// Launch
// ---------------------------------------------------------------------------
extern "C" void kernel_launch(void* const* d_in, const int* in_sizes, int n_in,
                              void* d_out, int out_size, void* d_ws, size_t ws_size,
                              hipStream_t stream) {
    (void)in_sizes; (void)n_in; (void)out_size; (void)ws_size;
    const float* x      = (const float*)d_in[0];
    const float* A_ssm  = (const float*)d_in[1];
    const float* B_ssm  = (const float*)d_in[2];
    const float* C_ssm  = (const float*)d_in[3];
    const float* D_ssm  = (const float*)d_in[4];
    const float* scale1 = (const float*)d_in[5];
    const float* scale2 = (const float*)d_in[6];
    const float* w1     = (const float*)d_in[7];
    const float* b1     = (const float*)d_in[8];
    const float* w2     = (const float*)d_in[9];
    const float* b2     = (const float*)d_in[10];
    float* out = (float*)d_out;

    char* cur = (char*)d_ws;
    auto carve = [&](size_t bytes) -> char* {
        char* p = cur;
        cur += (bytes + 255) & ~(size_t)255;
        return p;
    };
    float*          U    = (float*)carve((size_t)TOKENS * DMODEL * 4);                // 32MB
    float*          Y    = (float*)carve((size_t)TOKENS * DMODEL * 4);                // 32MB
    float*          X2   = (float*)carve((size_t)TOKENS * DMODEL * 4);                // 32MB
    float*          Send = (float*)carve((size_t)BATCH * NCHUNK * NSTATE * DMODEL * 4); // 8MB
    float*          Sin  = (float*)carve((size_t)BATCH * NCHUNK * NSTATE * DMODEL * 4); // 8MB
    unsigned short* Hb   = (unsigned short*)carve((size_t)TOKENS * DMODEL * 2);       // 16MB
    unsigned short* Gb   = (unsigned short*)carve((size_t)TOKENS * DFF * 2);          // 64MB
    unsigned short* W1t  = (unsigned short*)carve((size_t)DMODEL * DFF * 2);          // 8MB
    unsigned short* W2t  = (unsigned short*)carve((size_t)DFF * DMODEL * 2);          // 8MB

    // Weight convert + transpose to bf16 (one-time each call, cheap)
    conv_transpose_kernel<<<(DMODEL * DFF) / 256, 256, 0, stream>>>(w1, W1t, DMODEL, DFF);
    conv_transpose_kernel<<<(DFF * DMODEL) / 256, 256, 0, stream>>>(w2, W2t, DFF, DMODEL);

    // u = rmsnorm(x, scale1)
    rmsnorm_kernel<0><<<TOKENS, 256, 0, stream>>>(x, scale1, U, nullptr);

    // chunked SSM scan
    scan_phase1_kernel<<<BATCH * NCHUNK * (DMODEL / 256), 256, 0, stream>>>(
        U, A_ssm, B_ssm, C_ssm, Y, Send);
    scan_phase2_kernel<<<(BATCH * NSTATE * DMODEL) / 256, 256, 0, stream>>>(
        A_ssm, Send, Sin);
    scan_phase3_kernel<<<BATCH * NCHUNK * (DMODEL / 256), 256, 0, stream>>>(
        x, U, Y, Sin, A_ssm, C_ssm, D_ssm, X2);

    // h = rmsnorm(x2, scale2) -> bf16
    rmsnorm_kernel<1><<<TOKENS, 256, 0, stream>>>(X2, scale2, nullptr, Hb);

    // FFN: GEMM1 (+bias, GELU) -> bf16 ; GEMM2 (+bias, +x2 residual) -> f32 out
    gemm_bf16_kernel<0><<<dim3(TOKENS / 128, DFF / 128), 256, 0, stream>>>(
        Hb, W1t, b1, nullptr, (void*)Gb, TOKENS, DFF, DMODEL);
    gemm_bf16_kernel<1><<<dim3(TOKENS / 128, DMODEL / 128), 256, 0, stream>>>(
        Gb, W2t, b2, X2, (void*)out, TOKENS, DMODEL, DFF);
}
